// MultiHeadGATEAULayer_27032524161352
// MI455X (gfx1250) — matrix operand order repaired and checked by reading, not compiled, not run on hardware
//
#include <hip/hip_runtime.h>
#include <hip/hip_bf16.h>

typedef __attribute__((ext_vector_type(16))) __bf16 v16bf;
typedef __attribute__((ext_vector_type(8)))  float  v8f;

#define WMMA_BF16(a,b,c) \
  __builtin_amdgcn_wmma_f32_16x16x32_bf16(false,(a),false,(b),(short)0,(c),false,false)

constexpr int NNODES = 50000;
constexpr int NEDGES = 800000;
constexpr int D      = 128;
constexpr int NH     = 8;
constexpr int PACKSZ = 128 * 128;   // bf16 elements per packed weight matrix
constexpr int ETILES = 4;           // 16-edge tiles per block in the edge kernel

// ---------- order-preserving float <-> uint for atomic max ----------
__device__ __forceinline__ unsigned f2ord(float f) {
  unsigned u = __float_as_uint(f);
  return (u & 0x80000000u) ? ~u : (u | 0x80000000u);
}
__device__ __forceinline__ float ord2f(unsigned u) {
  u = (u & 0x80000000u) ? (u & 0x7fffffffu) : ~u;
  return __uint_as_float(u);
}

// ---------- CDNA5 async global->LDS copy (ASYNCcnt-tracked) ----------
__device__ __forceinline__ void async_ld_b128(unsigned lds_off, const float* gaddr) {
  asm volatile("global_load_async_to_lds_b128 %0, %1, off"
               :: "v"(lds_off), "v"(gaddr) : "memory");
}
__device__ __forceinline__ void wait_async_le2() {
  asm volatile("s_wait_asynccnt 0x2" ::: "memory");
}
__device__ __forceinline__ void wait_async_0() {
  asm volatile("s_wait_asynccnt 0x0" ::: "memory");
}

// ---------- operand builders ----------
// A operand (16x32 bf16): lane<16 -> row=lane, K = kb+0..7 then kb+16..23
//                         lane>=16 -> row=lane-16, K = kb+8..15 then kb+24..31
__device__ __forceinline__ void put4(v16bf& v, int i, float4 f) {
  v[i+0] = (__bf16)f.x; v[i+1] = (__bf16)f.y;
  v[i+2] = (__bf16)f.z; v[i+3] = (__bf16)f.w;
}

__device__ __forceinline__ v16bf load_a_f32(const float* A, int row, int ktbase, int lane) {
  int hi = (lane >> 4) & 1;
  const float4* p = (const float4*)(A + (size_t)row * D + ktbase + hi * 8);
  v16bf a;
  put4(a, 0,  p[0]);  put4(a, 4,  p[1]);   // K = kb .. kb+7
  put4(a, 8,  p[4]);  put4(a, 12, p[5]);   // K = kb+16 .. kb+23
  return a;
}

__device__ __forceinline__ v16bf load_a_sum(const float* A0, const float* A1,
                                            int row, int ktbase, int lane) {
  int hi = (lane >> 4) & 1;
  size_t off = (size_t)row * D + ktbase + hi * 8;
  const float* p0 = A0 + off;
  const float* p1 = A1 + off;
  v16bf a;
#pragma unroll
  for (int i = 0; i < 8; i++)  a[i]     = (__bf16)(p0[i]      + p1[i]);
#pragma unroll
  for (int i = 0; i < 8; i++)  a[8 + i] = (__bf16)(p0[16 + i] + p1[16 + i]);
  return a;
}

// build A-operand from an fp32 16x128 tile staged in LDS
__device__ __forceinline__ v16bf load_a_stagef32(const float* S, int kt, int lane) {
  int row = lane & 15;
  int kb  = kt * 32 + ((lane >> 4) & 1) * 8;
  const float* p = S + row * D + kb;
  v16bf a;
#pragma unroll
  for (int i = 0; i < 8; i++) a[i]     = (__bf16)p[i];
#pragma unroll
  for (int i = 0; i < 8; i++) a[8 + i] = (__bf16)p[16 + i];
  return a;
}

// build A-operand from a bf16 16x128 tile in LDS
__device__ __forceinline__ v16bf load_a_lds(const __bf16* L, int kt, int lane) {
  int row = lane & 15;
  int kb  = kt * 32 + ((lane >> 4) & 1) * 8;
  const __bf16* p = L + row * D + kb;
  v16bf a;
#pragma unroll
  for (int i = 0; i < 8; i++) a[i]     = p[i];
#pragma unroll
  for (int i = 0; i < 8; i++) a[8 + i] = p[16 + i];
  return a;
}

// B operand (32x16 bf16): lane<16 -> col=lane, K=0..7,16..23 of tile; lane>=16 -> K=8..15,24..31
__device__ __forceinline__ v16bf load_b(const __bf16* pack, int nt, int kt, int lane) {
  return *(const v16bf*)(pack + ((((size_t)nt * 4 + kt) * 32 + lane) << 4));
}

// ---------- kernel 0: prepack weights into B-operand layout (bf16) ----------
// mats: 0 Wv, 1 Wu, 2 We, 3 Wh, 4 Wg, 5 W0, 6 a_proj^T (pad to 16 cols), 7 Wout^T
__global__ void prepack_kernel(const float* Wv, const float* Wu, const float* We,
                               const float* Wh, const float* Wg, const float* W0,
                               const float* aP, const float* Wo, __bf16* packs) {
  int mat = blockIdx.x >> 3, nt = blockIdx.x & 7;
  int kt  = threadIdx.x >> 5, lane = threadIdx.x & 31;
  const float* W; int mode = 0;   // 0: B[k,n]=W[k*128+n]; 1: W[n*128+k]; 2: a_proj (n<8)
  switch (mat) {
    case 0: W = Wv; break;  case 1: W = Wu; break;
    case 2: W = We; break;  case 3: W = Wh; break;
    case 4: W = Wg; break;  case 5: W = W0; break;
    case 6: W = aP; mode = 2; break;
    default: W = Wo; mode = 1; break;
  }
  __bf16* out = packs + (size_t)mat * PACKSZ + ((((size_t)nt * 4 + kt) * 32 + lane) << 4);
  int n  = nt * 16 + (lane & 15);
  int kb = kt * 32 + ((lane >> 4) & 1) * 8;
#pragma unroll
  for (int e = 0; e < 16; e++) {
    int k = kb + (e < 8 ? e : e + 8);
    float v;
    if (mode == 0)      v = W[(size_t)k * D + n];
    else if (mode == 1) v = W[(size_t)n * D + k];
    else                v = (n < NH) ? W[(size_t)n * D + k] : 0.f;
    out[e] = (__bf16)v;
  }
}

// ---------- kernel 1: init softmax accumulators ----------
__global__ void init_kernel(unsigned* node_max, float* sum_exp, float* agg) {
  int idx = blockIdx.x * 256 + threadIdx.x;
  if (idx < NNODES * NH) { node_max[idx] = f2ord(-1.0e9f); sum_exp[idx] = 0.f; }
  if (idx < NNODES * D)  { agg[idx] = 0.f; }
}

// ---------- kernel 2: node-side fused GEMM: h_v,h_u,h_h,h_0 = X @ {Wv,Wu,Wh,W0} ----------
__global__ __launch_bounds__(256) void node_gemm_kernel(const float* X, const __bf16* packs,
                                                        float* h_v, float* h_u,
                                                        float* h_h, float* h_0) {
  int wave = threadIdx.x >> 5, lane = threadIdx.x & 31;
  int row0 = blockIdx.x * 16, nt = wave;
  int arow = row0 + (lane & 15);
  const __bf16* pv = packs + 0 * (size_t)PACKSZ;
  const __bf16* pu = packs + 1 * (size_t)PACKSZ;
  const __bf16* ph = packs + 3 * (size_t)PACKSZ;
  const __bf16* p0 = packs + 5 * (size_t)PACKSZ;
  v8f a0 = {}, a1 = {}, a2 = {}, a3 = {};
#pragma unroll
  for (int kt = 0; kt < 4; kt++) {
    v16bf a = load_a_f32(X, arow, kt * 32, lane);
    a0 = WMMA_BF16(a, load_b(pv, nt, kt, lane), a0);
    a1 = WMMA_BF16(a, load_b(pu, nt, kt, lane), a1);
    a2 = WMMA_BF16(a, load_b(ph, nt, kt, lane), a2);
    a3 = WMMA_BF16(a, load_b(p0, nt, kt, lane), a3);
  }
  int nn = lane & 15, hs = (lane >> 4) & 1;
#pragma unroll
  for (int r = 0; r < 8; r++) {
    size_t o = (size_t)(row0 + r + 8 * hs) * D + nt * 16 + nn;
    h_v[o] = a0[r]; h_u[o] = a1[r]; h_h[o] = a2[r]; h_0[o] = a3[r];
  }
}

// ---------- kernel 3: edge-side fused GEMM + gather + logits + atomic max ----------
// Double-buffered async global->LDS staging: each block walks ETILES consecutive
// 16-edge tiles; tile t+1 streams into LDS (ASYNCcnt) while tile t runs WMMAs.
__global__ __launch_bounds__(256) void edge_gemm_kernel(const float* EF, const __bf16* packs,
                                                        const float* h_v, const float* h_u,
                                                        const int* EI, const float* aB,
                                                        float* new_edge, __bf16* h_g,
                                                        float* scores, unsigned* node_max) {
  __shared__ float  stage[2][16 * D];   // fp32 edge-feature tiles (8KB each)
  __shared__ __bf16 lds_ne[16 * D];     // bf16 new_edge tile for the logits WMMA
  int tid  = threadIdx.x;
  int wave = tid >> 5, lane = tid & 31;
  int tile0 = blockIdx.x * ETILES;

  const __bf16* pe = packs + 2 * (size_t)PACKSZ;
  const __bf16* pg = packs + 4 * (size_t)PACKSZ;
  const __bf16* pa = packs + 6 * (size_t)PACKSZ;

  // each thread async-copies two b128 chunks of the 16x128 fp32 tile
  auto prefetch = [&](int it, int buf) {
    const float* g = EF + ((size_t)(tile0 + it) * 16) * D + tid * 4;
    unsigned l = (unsigned)(size_t)&stage[buf][tid * 4];
    async_ld_b128(l, g);
    async_ld_b128(l + 4096, g + 1024);
  };

  prefetch(0, 0);
#pragma unroll 1
  for (int it = 0; it < ETILES; ++it) {
    int buf = it & 1;
    if (it + 1 < ETILES) { prefetch(it + 1, buf ^ 1); wait_async_le2(); }
    else                 { wait_async_0(); }
    __syncthreads();                      // whole tile visible in LDS

    int row0 = (tile0 + it) * 16;
    int nt = wave;
    v8f accE = {}, accG = {};
#pragma unroll
    for (int kt = 0; kt < 4; kt++) {
      v16bf a = load_a_stagef32(stage[buf], kt, lane);
      accE = WMMA_BF16(a, load_b(pe, nt, kt, lane), accE);
      accG = WMMA_BF16(a, load_b(pg, nt, kt, lane), accG);
    }
    int nn = lane & 15, hs = (lane >> 4) & 1;
#pragma unroll
    for (int r = 0; r < 8; r++) {
      int e   = row0 + r + 8 * hs;
      int col = nt * 16 + nn;
      int t = EI[e];
      int s = EI[NEDGES + e];
      // new_edge = h_e + h_u[tgt] + h_v[src]  (gather tables resident in 192MB L2)
      float v = accE[r] + h_u[(size_t)t * D + col] + h_v[(size_t)s * D + col];
      new_edge[(size_t)e * D + col] = v;
      lds_ne[(r + 8 * hs) * D + col] = (__bf16)v;
      h_g[(size_t)e * D + col] = (__bf16)accG[r];
    }
    __syncthreads();                      // lds_ne complete
    if (wave == 0) {
      // logits[16 x 8] = new_edge[16x128] @ a_proj^T (zero-padded to 16 cols)
      v8f accL = {};
#pragma unroll
      for (int kt = 0; kt < 4; kt++) {
        v16bf a = load_a_lds(lds_ne, kt, lane);
        accL = WMMA_BF16(a, load_b(pa, 0, kt, lane), accL);
      }
      int head = lane & 15;
      if (head < NH) {
        float b = aB[head];
#pragma unroll
        for (int r = 0; r < 8; r++) {
          int e = row0 + r + 8 * hs;
          float sc = accL[r] + b;
          sc = (sc >= 0.f) ? sc : 0.2f * sc;        // leaky relu
          scores[(size_t)e * NH + head] = sc;
          int t = EI[e];
          atomicMax(&node_max[(size_t)t * NH + head], f2ord(sc));
        }
      }
    }
    __syncthreads();                      // protect lds_ne / stage reuse
  }
}

// ---------- kernel 4: exp + segment sum ----------
__global__ void exp_kernel(float* scores, const unsigned* node_max,
                           const int* EI, float* sum_exp) {
  int idx = blockIdx.x * 256 + threadIdx.x;
  if (idx >= NEDGES * NH) return;
  int e = idx >> 3, h = idx & 7;
  int t = EI[e];
  float m  = ord2f(node_max[(size_t)t * NH + h]);
  float ex = __expf(scores[idx] - m);
  scores[idx] = ex;
  atomicAdd(&sum_exp[(size_t)t * NH + h], ex);
}

// ---------- kernel 5: weighted message scatter-add (one wave per edge) ----------
__global__ __launch_bounds__(256) void scatter_kernel(const float* h_h, const __bf16* h_g,
                                                      const float* expsc, const float* sum_exp,
                                                      const int* EI, float* agg) {
  int gw   = (blockIdx.x * 256 + threadIdx.x) >> 5;  // global wave = edge id
  int lane = threadIdx.x & 31;
  if (gw >= NEDGES) return;
  int e = gw;
  int t = EI[e];
  int s = EI[NEDGES + e];
#pragma unroll
  for (int j = 0; j < 4; j++) {
    int col  = lane + 32 * j;
    int head = col >> 4;
    float alpha = expsc[(size_t)e * NH + head] /
                  (sum_exp[(size_t)t * NH + head] + 1e-10f);
    float v = h_h[(size_t)s * D + col] + (float)h_g[(size_t)e * D + col];
    atomicAdd(&agg[(size_t)t * D + col], v * alpha);
  }
}

// ---------- kernel 6: final GEMM: out = (h_0 + agg) @ Wout^T + b ----------
__global__ __launch_bounds__(256) void final_gemm_kernel(const float* h_0, const float* agg,
                                                         const __bf16* packs, const float* WoB,
                                                         float* out) {
  int wave = threadIdx.x >> 5, lane = threadIdx.x & 31;
  int row0 = blockIdx.x * 16, nt = wave;
  int arow = row0 + (lane & 15);
  const __bf16* pw = packs + 7 * (size_t)PACKSZ;
  v8f acc = {};
#pragma unroll
  for (int kt = 0; kt < 4; kt++) {
    v16bf a = load_a_sum(h_0, agg, arow, kt * 32, lane);
    acc = WMMA_BF16(a, load_b(pw, nt, kt, lane), acc);
  }
  int nn = lane & 15, hs = (lane >> 4) & 1;
#pragma unroll
  for (int r = 0; r < 8; r++) {
    int col = nt * 16 + nn;
    out[(size_t)(row0 + r + 8 * hs) * D + col] = acc[r] + WoB[col];
  }
}

// ---------- host launcher ----------
extern "C" void kernel_launch(void* const* d_in, const int* in_sizes, int n_in,
                              void* d_out, int out_size, void* d_ws, size_t ws_size,
                              hipStream_t stream) {
  const float* X   = (const float*)d_in[0];
  const float* EF  = (const float*)d_in[1];
  const float* Wv  = (const float*)d_in[2];
  const float* Wu  = (const float*)d_in[3];
  const float* We  = (const float*)d_in[4];
  const float* Wh  = (const float*)d_in[5];
  const float* Wg  = (const float*)d_in[6];
  const float* W0  = (const float*)d_in[7];
  const float* aW  = (const float*)d_in[8];
  const float* aB  = (const float*)d_in[9];
  const float* WoW = (const float*)d_in[10];
  const float* WoB = (const float*)d_in[11];
  const int*   EI  = (const int*)d_in[12];
  // d_in[13] edge_map is the identity permutation; unused.

  float* out_final = (float*)d_out;                       // [NNODES, 128]
  float* out_edge  = (float*)d_out + (size_t)NNODES * D;  // [NEDGES, 128]

  // workspace carve-out (256B aligned)
  char* w = (char*)d_ws;
  auto carve = [&](size_t bytes) -> char* {
    char* p = w; w += (bytes + 255) & ~(size_t)255; return p;
  };
  float*    h_v      = (float*)   carve((size_t)NNODES * D * 4);
  float*    h_u      = (float*)   carve((size_t)NNODES * D * 4);
  float*    h_h      = (float*)   carve((size_t)NNODES * D * 4);
  float*    h_0      = (float*)   carve((size_t)NNODES * D * 4);
  float*    agg      = (float*)   carve((size_t)NNODES * D * 4);
  float*    scores   = (float*)   carve((size_t)NEDGES * NH * 4);
  float*    sum_exp  = (float*)   carve((size_t)NNODES * NH * 4);
  unsigned* node_max = (unsigned*)carve((size_t)NNODES * NH * 4);
  __bf16*   h_g      = (__bf16*)  carve((size_t)NEDGES * D * 2);
  __bf16*   packs    = (__bf16*)  carve((size_t)8 * PACKSZ * 2);

  prepack_kernel<<<64, 128, 0, stream>>>(Wv, Wu, We, Wh, Wg, W0, aW, WoW, packs);
  init_kernel<<<(NNODES * D + 255) / 256, 256, 0, stream>>>(node_max, sum_exp, agg);
  node_gemm_kernel<<<NNODES / 16, 256, 0, stream>>>(X, packs, h_v, h_u, h_h, h_0);
  edge_gemm_kernel<<<NEDGES / (16 * ETILES), 256, 0, stream>>>(EF, packs, h_v, h_u, EI, aB,
                                                               out_edge, h_g, scores, node_max);
  exp_kernel<<<(NEDGES * NH + 255) / 256, 256, 0, stream>>>(scores, node_max, EI, sum_exp);
  scatter_kernel<<<(NEDGES * 32 + 255) / 256, 256, 0, stream>>>(h_h, h_g, scores, sum_exp,
                                                                EI, agg);
  final_gemm_kernel<<<NNODES / 16, 256, 0, stream>>>(h_0, agg, packs, WoB, out_final);
}
